// DeepHamSupervised_85040352461331
// MI455X (gfx1250) — compile-verified, matrix operand order checked by reading, not run on hardware
//
#include <hip/hip_runtime.h>
#include <hip/hip_bf16.h>
#include <math.h>

typedef __attribute__((ext_vector_type(16))) _Float16 v16h;
typedef __attribute__((ext_vector_type(8)))  _Float16 v8h;
typedef __attribute__((ext_vector_type(8)))  float    v8f;

#define CAT16(lo, hi) __builtin_shufflevector((lo), (hi), 0,1,2,3,4,5,6,7,8,9,10,11,12,13,14,15)

// ---------------------------------------------------------------------------
// helpers
// ---------------------------------------------------------------------------
__device__ __forceinline__ float wave_reduce_sum(float v) {
#pragma unroll
    for (int off = 16; off > 0; off >>= 1)
        v += __shfl_xor(v, off, 32);
    return v;
}

__device__ __forceinline__ void atomicMaxF(float* addr, float val) {
    unsigned int* p = (unsigned int*)addr;
    unsigned int old = *p;
    while (true) {
        float cur = __uint_as_float(old);
        if (cur >= val) break;
        unsigned int assumed = old;
        old = atomicCAS(p, assumed, __float_as_uint(val));
        if (old == assumed) break;
    }
}

__device__ __forceinline__ v16h load_frag(const _Float16* __restrict__ base,
                                          unsigned off)
{
    v8h lo = *(const v8h*)(base + off);
    v8h hi = *(const v8h*)(base + off + 16);
    return CAT16(lo, hi);
}

// ---------------------------------------------------------------------------
// one-time prep: W[K][N] fp32  ->  Wt[N][K] f16  (B fragments become rows)
// ---------------------------------------------------------------------------
__global__ void transpose_to_f16(const float* __restrict__ W,
                                 _Float16* __restrict__ Wt, int K, int N)
{
    int idx = blockIdx.x * blockDim.x + threadIdx.x;
    if (idx >= K * N) return;
    int n = idx / K;
    int k = idx - n * K;                 // output contiguous in k
    Wt[idx] = (_Float16)W[(size_t)k * N + n];
}

// fp32 -> f16 elementwise (activations)
__global__ void cvt_to_f16(const float* __restrict__ X,
                           _Float16* __restrict__ Y, int n)
{
    int i = blockIdx.x * blockDim.x + threadIdx.x;
    if (i < n) Y[i] = (_Float16)X[i];
}

// ---------------------------------------------------------------------------
// WMMA GEMM, all-f16 operands, f32 accumulate.
//   C1[M,N] = act(A16[M,K] @ W1t[N,K]^T + bias1); optionally C2 = A @ W2t^T.
// One wave computes a 16(M) x 64(N) strip: 4 N-tiles sharing one A fragment,
// 8 independent v_wmma per 32-wide k-step. Requires N % 64 == 0, K % 32 == 0.
// All offsets kept in 32 bits (matrices < 4G elements) so the compiler can
// use uniform-SGPR-base + VGPR-offset addressing; B fragments are loaded into
// distinct register arrays before the WMMA burst so loads clause together.
// Fragment mapping per CDNA5 ISA 7.12.2:
//   lane = {m15, khalf}; frag element j -> K = kk + khalf*8 + {j, j+8 for j>=8}
//   => two contiguous 8-half (16B) loads at kof and kof+16.
// act: 0 = identity, 1 = leaky_relu(0.1)
// ---------------------------------------------------------------------------
__global__ __launch_bounds__(32) void gemm_wmma_f16(
    const _Float16* __restrict__ A16, const _Float16* __restrict__ W1t,
    const _Float16* __restrict__ W2t, const float* __restrict__ bias1,
    float* __restrict__ C1, float* __restrict__ C2,
    int M, int K, int N, int act)
{
    const int lane  = threadIdx.x & 31;
    const int m15   = lane & 15;
    const int khalf = lane >> 4;
    const int tileM = blockIdx.y;
    const int ng    = blockIdx.x;                 // group of 4 N-tiles
    const int row   = tileM * 16 + m15;

    v8f acc1[4] = {{}, {}, {}, {}};
    v8f acc2[4] = {{}, {}, {}, {}};

    const unsigned aoff0 = (unsigned)(row * K) + (unsigned)(khalf * 8);
    unsigned woff[4];
#pragma unroll
    for (int t = 0; t < 4; ++t)
        woff[t] = (unsigned)((ng * 64 + t * 16 + m15) * K) + (unsigned)(khalf * 8);

    // software pipeline on the A fragment
    v16h a_cur = load_frag(A16, aoff0);

    for (int kk = 0; kk < K; kk += 32) {
        const unsigned kof = (unsigned)kk;

        // ---- gather all B fragments for this k-step into distinct regs ----
        v16h b1f[4];
#pragma unroll
        for (int t = 0; t < 4; ++t)
            b1f[t] = load_frag(W1t, woff[t] + kof);

        v16h b2f[4];
        if (W2t) {
#pragma unroll
            for (int t = 0; t < 4; ++t)
                b2f[t] = load_frag(W2t, woff[t] + kof);
        }

        // ---- prefetch next A fragment while WMMAs run ----
        v16h a_use = a_cur;
        if (kk + 32 < K)
            a_cur = load_frag(A16, aoff0 + (unsigned)(kk + 32));

        // ---- 8 independent WMMAs ----
#pragma unroll
        for (int t = 0; t < 4; ++t)
            acc1[t] = __builtin_amdgcn_wmma_f32_16x16x32_f16(
                false, a_use, false, b1f[t], (short)0, acc1[t], false, false);
        if (W2t) {
#pragma unroll
            for (int t = 0; t < 4; ++t)
                acc2[t] = __builtin_amdgcn_wmma_f32_16x16x32_f16(
                    false, a_use, false, b2f[t], (short)0, acc2[t], false, false);
        }
    }

    // epilogue: C/D layout — VGPR i -> M = i (+8 for upper lane half), N = m15
    const int mofs = (lane >= 16) ? 8 : 0;
#pragma unroll
    for (int t = 0; t < 4; ++t) {
        int col  = ng * 64 + t * 16 + m15;
        float bv = bias1 ? bias1[col] : 0.f;
#pragma unroll
        for (int i = 0; i < 8; ++i) {
            unsigned o = (unsigned)((tileM * 16 + i + mofs) * N + col);
            float v = acc1[t][i] + bv;
            if (act == 1) v = (v > 0.f) ? v : 0.1f * v;
            C1[o] = v;
            if (W2t) C2[o] = acc2[t][i];
        }
    }
}

// ---------------------------------------------------------------------------
// per-layer init: agg = 0, emax = -inf, denom = 0
// ---------------------------------------------------------------------------
__global__ void init_layer(float* __restrict__ agg, float* __restrict__ emax,
                           float* __restrict__ denom, int ND, int N)
{
    int i = blockIdx.x * blockDim.x + threadIdx.x;
    if (i < ND) agg[i] = 0.f;
    if (i < N) { emax[i] = -INFINITY; denom[i] = 0.f; }
}

// ---------------------------------------------------------------------------
// e[edge] = att . leaky_relu(xl[src] + xr[dst], 0.2); emax[dst] = max(...)
// ---------------------------------------------------------------------------
__global__ __launch_bounds__(256) void edge_logits_kernel(
    const float* __restrict__ xl, const float* __restrict__ xr,
    const float* __restrict__ att, const int* __restrict__ src,
    const int* __restrict__ dst, float* __restrict__ e,
    float* __restrict__ emax, int E, int D)
{
    int edge = blockIdx.x * 8 + (threadIdx.x >> 5);
    int lane = threadIdx.x & 31;
    if (edge >= E) return;
    int s = src[edge], d = dst[edge];
    const float* pl = xl + (size_t)s * D;
    const float* pr = xr + (size_t)d * D;
    float part = 0.f;
    for (int f = lane; f < D; f += 32) {
        float v = pl[f] + pr[f];
        v = (v > 0.f) ? v : 0.2f * v;
        part += att[f] * v;
    }
    part = wave_reduce_sum(part);
    if (lane == 0) {
        e[edge] = part;
        atomicMaxF(&emax[d], part);
    }
}

// ---------------------------------------------------------------------------
// w = exp(e - emax[dst]); denom[dst] += w   (w overwrites e)
// ---------------------------------------------------------------------------
__global__ void edge_softmax_num(const float* __restrict__ emax,
                                 float* __restrict__ e,
                                 float* __restrict__ denom,
                                 const int* __restrict__ dst, int E)
{
    int i = blockIdx.x * blockDim.x + threadIdx.x;
    if (i >= E) return;
    int d = dst[i];
    float w = expf(e[i] - emax[d]);
    e[i] = w;
    atomicAdd(&denom[d], w);
}

// ---------------------------------------------------------------------------
// agg[dst] += (w / denom[dst]) * xl[src]   (one wave per edge)
// ---------------------------------------------------------------------------
__global__ __launch_bounds__(256) void edge_scatter_kernel(
    const float* __restrict__ w, const float* __restrict__ denom,
    const float* __restrict__ xl, const int* __restrict__ src,
    const int* __restrict__ dst, float* __restrict__ agg, int E, int D)
{
    int edge = blockIdx.x * 8 + (threadIdx.x >> 5);
    int lane = threadIdx.x & 31;
    if (edge >= E) return;
    int s = src[edge], d = dst[edge];
    float alpha = w[edge] / (denom[d] + 1e-16f);
    const float* pl = xl + (size_t)s * D;
    float* pa = agg + (size_t)d * D;
    for (int f = lane; f < D; f += 32)
        atomicAdd(&pa[f], alpha * pl[f]);
}

// ---------------------------------------------------------------------------
// h = tanh(agg + bias)
// ---------------------------------------------------------------------------
__global__ void node_tanh_bias(const float* __restrict__ agg,
                               const float* __restrict__ bias,
                               float* __restrict__ h, int total, int D)
{
    int i = blockIdx.x * blockDim.x + threadIdx.x;
    if (i >= total) return;
    h[i] = tanhf(agg[i] + bias[i & (D - 1)]);   // D == 512 (power of two)
}

// ---------------------------------------------------------------------------
// logits[node] = h[node,:] . A3 + c3   (one wave per node, D = 256)
// ---------------------------------------------------------------------------
__global__ __launch_bounds__(256) void final_dot_kernel(
    const float* __restrict__ h, const float* __restrict__ A3,
    const float* __restrict__ c3, float* __restrict__ logits, int N, int D)
{
    int node = blockIdx.x * 8 + (threadIdx.x >> 5);
    int lane = threadIdx.x & 31;
    if (node >= N) return;
    const float* p = h + (size_t)node * D;
    float part = 0.f;
    for (int f = lane; f < D; f += 32)
        part += p[f] * A3[f];
    part = wave_reduce_sum(part);
    if (lane == 0) logits[node] = part + c3[0];
}

// ---------------------------------------------------------------------------
// softmax over all N logits (single block, 3 phases)
// ---------------------------------------------------------------------------
__global__ __launch_bounds__(1024) void softmax_nodes(
    const float* __restrict__ logits, float* __restrict__ out, int N)
{
    __shared__ float red[1024];
    __shared__ float s_max, s_sum;
    int tid = threadIdx.x;

    float m = -INFINITY;
    for (int i = tid; i < N; i += 1024) m = fmaxf(m, logits[i]);
    red[tid] = m; __syncthreads();
    for (int s = 512; s > 0; s >>= 1) {
        if (tid < s) red[tid] = fmaxf(red[tid], red[tid + s]);
        __syncthreads();
    }
    if (tid == 0) s_max = red[0];
    __syncthreads();

    float sum = 0.f;
    for (int i = tid; i < N; i += 1024) sum += expf(logits[i] - s_max);
    red[tid] = sum; __syncthreads();
    for (int s = 512; s > 0; s >>= 1) {
        if (tid < s) red[tid] += red[tid + s];
        __syncthreads();
    }
    if (tid == 0) s_sum = red[0];
    __syncthreads();

    float inv = 1.f / s_sum;
    for (int i = tid; i < N; i += 1024)
        out[i] = expf(logits[i] - s_max) * inv;
}

// ---------------------------------------------------------------------------
// host entry
// ---------------------------------------------------------------------------
extern "C" void kernel_launch(void* const* d_in, const int* in_sizes, int n_in,
                              void* d_out, int out_size, void* d_ws, size_t ws_size,
                              hipStream_t stream)
{
    (void)n_in; (void)out_size; (void)ws_size;
    const int DI = 128, DE = 512, DH = 256;

    const float* x  = (const float*)d_in[0];
    const int*   ei = (const int*)d_in[1];
    const int N = in_sizes[0] / DI;
    const int E = in_sizes[1] / 2;
    const int* src = ei;
    const int* dst = ei + E;

    const float* Wl[3]  = { (const float*)d_in[2],  (const float*)d_in[6],  (const float*)d_in[10] };
    const float* Wr[3]  = { (const float*)d_in[3],  (const float*)d_in[7],  (const float*)d_in[11] };
    const float* att[3] = { (const float*)d_in[4],  (const float*)d_in[8],  (const float*)d_in[12] };
    const float* bia[3] = { (const float*)d_in[5],  (const float*)d_in[9],  (const float*)d_in[13] };
    const float* A1 = (const float*)d_in[14]; const float* c1 = (const float*)d_in[15];
    const float* A2 = (const float*)d_in[16]; const float* c2 = (const float*)d_in[17];
    const float* A3 = (const float*)d_in[18]; const float* c3 = (const float*)d_in[19];

    // ---------------- workspace layout ----------------
    float* ws = (float*)d_ws;
    const size_t nd = (size_t)N * DE;
    float* h      = ws; ws += nd;
    float* xl     = ws; ws += nd;
    float* xr     = ws; ws += nd;
    float* agg    = ws; ws += nd;
    float* ebuf   = ws; ws += E;
    float* emax   = ws; ws += N;
    float* denom  = ws; ws += N;
    float* hid1   = ws; ws += (size_t)N * DH;
    float* hid2   = ws; ws += (size_t)N * DH;
    float* logits = ws; ws += N;

    _Float16* hp = (_Float16*)ws;
    _Float16* x16    = hp; hp += (size_t)N * DI;
    _Float16* h16    = hp; hp += (size_t)N * DE;
    _Float16* hid116 = hp; hp += (size_t)N * DH;
    _Float16* Wlt[3], *Wrt[3];
    Wlt[0] = hp; hp += (size_t)DI * DE;    // [DE][DI]
    Wrt[0] = hp; hp += (size_t)DI * DE;
    Wlt[1] = hp; hp += (size_t)DE * DE;
    Wrt[1] = hp; hp += (size_t)DE * DE;
    Wlt[2] = hp; hp += (size_t)DE * DE;
    Wrt[2] = hp; hp += (size_t)DE * DE;
    _Float16* A1t = hp; hp += (size_t)DE * DH;   // [DH][DE]
    _Float16* A2t = hp; hp += (size_t)DH * DH;   // [DH][DH]

    const int Mt = N / 16;                       // 625
    const dim3 gGemmE(DE / 64, Mt);              // 4 N-tiles per wave
    const dim3 gGemmH(DH / 64, Mt);
    const int ndBlocks = (int)((nd + 255) / 256);
    const int eBlocks8 = (E + 7) / 8;
    const int eBlocks  = (E + 255) / 256;
    auto blk = [](size_t n) { return (unsigned)((n + 255) / 256); };

    // ---------------- one-time prep: f16 transposed weights, f16 x ----------
    transpose_to_f16<<<blk((size_t)DI * DE), 256, 0, stream>>>(Wl[0], Wlt[0], DI, DE);
    transpose_to_f16<<<blk((size_t)DI * DE), 256, 0, stream>>>(Wr[0], Wrt[0], DI, DE);
    transpose_to_f16<<<blk((size_t)DE * DE), 256, 0, stream>>>(Wl[1], Wlt[1], DE, DE);
    transpose_to_f16<<<blk((size_t)DE * DE), 256, 0, stream>>>(Wr[1], Wrt[1], DE, DE);
    transpose_to_f16<<<blk((size_t)DE * DE), 256, 0, stream>>>(Wl[2], Wlt[2], DE, DE);
    transpose_to_f16<<<blk((size_t)DE * DE), 256, 0, stream>>>(Wr[2], Wrt[2], DE, DE);
    transpose_to_f16<<<blk((size_t)DE * DH), 256, 0, stream>>>(A1, A1t, DE, DH);
    transpose_to_f16<<<blk((size_t)DH * DH), 256, 0, stream>>>(A2, A2t, DH, DH);
    cvt_to_f16<<<blk((size_t)N * DI), 256, 0, stream>>>(x, x16, N * DI);

    // ---------------- GATv2 layers ----------------
    for (int l = 0; l < 3; ++l) {
        const _Float16* in16 = (l == 0) ? x16 : h16;
        const int K = (l == 0) ? DI : DE;
        gemm_wmma_f16<<<gGemmE, 32, 0, stream>>>(in16, Wlt[l], Wrt[l], nullptr,
                                                 xl, xr, N, K, DE, 0);
        init_layer<<<ndBlocks, 256, 0, stream>>>(agg, emax, denom, (int)nd, N);
        edge_logits_kernel<<<eBlocks8, 256, 0, stream>>>(xl, xr, att[l], src, dst,
                                                         ebuf, emax, E, DE);
        edge_softmax_num<<<eBlocks, 256, 0, stream>>>(emax, ebuf, denom, dst, E);
        edge_scatter_kernel<<<eBlocks8, 256, 0, stream>>>(ebuf, denom, xl, src, dst,
                                                          agg, E, DE);
        node_tanh_bias<<<ndBlocks, 256, 0, stream>>>(agg, bia[l], h, (int)nd, DE);
        cvt_to_f16<<<ndBlocks, 256, 0, stream>>>(h, h16, (int)nd);
    }

    // ---------------- MLP head ----------------
    gemm_wmma_f16<<<gGemmH, 32, 0, stream>>>(h16, A1t, nullptr, c1,
                                             hid1, nullptr, N, DE, DH, 1);
    cvt_to_f16<<<blk((size_t)N * DH), 256, 0, stream>>>(hid1, hid116, N * DH);
    gemm_wmma_f16<<<gGemmH, 32, 0, stream>>>(hid116, A2t, nullptr, c2,
                                             hid2, nullptr, N, DH, DH, 1);
    final_dot_kernel<<<(N + 7) / 8, 256, 0, stream>>>(hid2, A3, c3, logits, N, DH);
    softmax_nodes<<<1, 1024, 0, stream>>>(logits, (float*)d_out, N);
}
